// Model_18863496364288
// MI455X (gfx1250) — matrix operand order repaired and checked by reading, not compiled, not run on hardware
//
#include <hip/hip_runtime.h>
#include <stdint.h>

// ---------------- model constants ----------------
#define BN   8      // batch
#define TT   16     // time steps
#define NC   1024   // CODE_NUM
#define CS   48     // CODE_SIZE
#define GS   32     // GRAPH_SIZE == T_ATTN
#define HID  150    // hidden size
#define HP   160    // hidden padded to 10 wmma col-tiles
#define NCOL (BN*CS)   // 384 columns of the adjacency GEMM
#define OUTN 1024
#define NEGINF (-1.0e30f)

typedef __attribute__((ext_vector_type(16))) __bf16 v16bf;
typedef __attribute__((ext_vector_type(8)))  float  v8f;

union FragBF { v16bf v; uint4 q[2]; };

static __device__ __forceinline__ unsigned short f2bf(float f) {
  unsigned u = __float_as_uint(f);
  u += 0x7fffu + ((u >> 16) & 1u);      // round-to-nearest-even
  return (unsigned short)(u >> 16);
}
// monotonic unsigned encoding of float for atomicMax-based max-pool
static __device__ __forceinline__ unsigned encf(float f) {
  unsigned u = __float_as_uint(f);
  return (u & 0x80000000u) ? ~u : (u | 0x80000000u);
}
static __device__ __forceinline__ float decf(unsigned e) {
  return (e & 0x80000000u) ? __uint_as_float(e & 0x7fffffffu) : __uint_as_float(~e);
}

// ---------------- tiny utility kernels ----------------
__global__ void k_zero(float* p, int n) {
  int i = blockIdx.x * 256 + threadIdx.x;
  if (i < n) p[i] = 0.f;
}
__global__ void k_cvt_bf16(const float* __restrict__ s, unsigned short* __restrict__ d, int n) {
  int i = blockIdx.x * 256 + threadIdx.x;
  if (i < n) d[i] = f2bf(s[i]);
}
__global__ void k_clear_om(unsigned* a, unsigned* b) {
  int i = blockIdx.x * 256 + threadIdx.x;
  if (i < BN * HP) { a[i] = 0u; b[i] = 0u; }
}

// build transposed, mask-scaled bf16 embedding matrices for the adjacency GEMM:
// cet[col=b*48+c][n] = code_x[b,t,n] * c_emb[n,c];  net likewise with neighbors/n_emb
__global__ void k_mask_embed(const float* __restrict__ code_x, const float* __restrict__ neigh,
                             const float* __restrict__ c_emb, const float* __restrict__ n_emb,
                             unsigned short* __restrict__ cet, unsigned short* __restrict__ net,
                             int t) {
  int i = blockIdx.x * 256 + threadIdx.x;          // over NCOL*NC
  if (i >= NCOL * NC) return;
  int col = i >> 10, n = i & (NC - 1);
  int b = col / CS, c = col % CS;
  float cm = code_x[(b * TT + t) * NC + n];
  float nm = neigh [(b * TT + t) * NC + n];
  cet[i] = f2bf(cm * c_emb[n * CS + c]);
  net[i] = f2bf(nm * n_emb[n * CS + c]);
}

// ---------------- bf16 WMMA GEMM: C[1024,384] = A[1024,1024] * Bt[384,1024]^T ----------------
// one wave per 16x64 output strip (A fragment reused across 4 column tiles)
__global__ void __launch_bounds__(32)
k_gemm_adj(const unsigned short* __restrict__ A, const unsigned short* __restrict__ Bt,
           float* __restrict__ C) {
  int lane = threadIdx.x & 31;
  int half = lane >> 4, lm = lane & 15;
  int m0 = blockIdx.x * 16;
  int n0 = blockIdx.y * 64;
  v8f acc[4] = {};
  const unsigned short* arow = A + (size_t)(m0 + lm) * NC;
  for (int k0 = 0; k0 < NC; k0 += 32) {
    FragBF af;
    af.q[0] = *(const uint4*)(arow + k0 + half * 16);
    af.q[1] = *(const uint4*)(arow + k0 + half * 16 + 8);
#pragma unroll
    for (int j = 0; j < 4; ++j) {
      FragBF bf;
      const unsigned short* bp = Bt + (size_t)(n0 + j * 16 + lm) * NC + k0 + half * 16;
      bf.q[0] = ((const uint4*)bp)[0];
      bf.q[1] = ((const uint4*)bp)[1];
      acc[j] = __builtin_amdgcn_wmma_f32_16x16x32_bf16(false, af.v, false, bf.v,
                                                       (short)0, acc[j], false, false);
    }
  }
#pragma unroll
  for (int j = 0; j < 4; ++j)
#pragma unroll
    for (int r = 0; r < 8; ++r)
      C[(size_t)(m0 + r + half * 8) * NCOL + n0 + j * 16 + lm] = acc[j][r];
}

// ---------------- GraphLayer dense: co/no = leaky_relu(x @ Wg + bg) ----------------
__global__ void k_co_no(const float* __restrict__ code_x, const float* __restrict__ neigh,
                        const float* __restrict__ c_emb, const float* __restrict__ n_emb,
                        const float* __restrict__ adj_ce, const float* __restrict__ adj_ne,
                        const float* __restrict__ Wg, const float* __restrict__ bg,
                        float* __restrict__ co, float* __restrict__ no_cur, int t) {
  int i = blockIdx.x * 256 + threadIdx.x;          // b*NC + n
  if (i >= BN * NC) return;
  int b = i >> 10, n = i & (NC - 1);
  float cm = code_x[(b * TT + t) * NC + n];
  float nm = neigh [(b * TT + t) * NC + n];
  float xc[CS], xn[CS];
  int col0 = b * CS;
  for (int c = 0; c < CS; ++c) {
    float s = adj_ce[(size_t)n * NCOL + col0 + c] + adj_ne[(size_t)n * NCOL + col0 + c];
    xc[c] = cm * (c_emb[n * CS + c] + s);   // ce + c*adj_ce + c*adj_ne
    xn[c] = nm * (n_emb[n * CS + c] + s);   // ne + nb*adj_ne + nb*adj_ce
  }
  for (int g = 0; g < GS; ++g) {
    float a0 = bg[g], a1 = bg[g];
    for (int c = 0; c < CS; ++c) {
      float w = Wg[c * GS + g];
      a0 += xc[c] * w; a1 += xn[c] * w;
    }
    co    [(size_t)i * GS + g] = a0 > 0.f ? a0 : 0.01f * a0;
    no_cur[(size_t)i * GS + g] = a1 > 0.f ? a1 : 0.01f * a1;
  }
}

// ---------------- q/k/v projections (only for active m2|m3 codes) ----------------
__global__ void k_qkv(const float* __restrict__ divided, const float* __restrict__ no_prev,
                      const float* __restrict__ u_emb, const float* __restrict__ co,
                      const float* __restrict__ Wq, const float* __restrict__ bq,
                      const float* __restrict__ Wk, const float* __restrict__ bk,
                      const float* __restrict__ Wv, const float* __restrict__ bv,
                      unsigned short* __restrict__ qb, unsigned short* __restrict__ kb,
                      unsigned short* __restrict__ vtb, int t) {
  int i = blockIdx.x * 256 + threadIdx.x;          // b*NC + n
  if (i >= BN * NC) return;
  int b = i >> 10, n = i & (NC - 1);
  const float* dv = divided + ((size_t)(b * TT + t) * NC + n) * 3;
  float m2 = dv[1], m3 = dv[2];
  if ((m2 + m3) <= 0.f) return;                    // masked keys/queries never contribute
  float qr[GS], cr[GS];
  for (int j = 0; j < GS; ++j) {
    qr[j] = (m2 > 0.f) ? no_prev[(size_t)i * GS + j]
                       : ((m3 > 0.f) ? u_emb[n * GS + j] : 0.f);
    cr[j] = co[(size_t)i * GS + j];
  }
  for (int a = 0; a < GS; ++a) {
    float q = bq[a], k = bk[a];
    for (int j = 0; j < GS; ++j) { q += qr[j] * Wq[j * GS + a]; k += cr[j] * Wk[j * GS + a]; }
    qb[(size_t)i * GS + a] = f2bf(q);
    kb[(size_t)i * GS + a] = f2bf(k);
  }
  for (int h = 0; h < HID; ++h) {
    float v = bv[h];
    for (int j = 0; j < GS; ++j) v += qr[j] * Wv[j * HID + h];
    vtb[((size_t)b * HP + h) * NC + n] = f2bf(v);  // transposed for B-fragments
  }
}

// ---------------- GRUCell on persistent (m1) codes ----------------
__global__ void __launch_bounds__(HP)
k_gru(const float* __restrict__ divided, const float* __restrict__ co,
      const float* __restrict__ h_prev,
      const float* __restrict__ Wi, const float* __restrict__ bi,
      const float* __restrict__ Wh, const float* __restrict__ bh,
      float* __restrict__ h_new, unsigned* __restrict__ om1e, int t) {
  int i = blockIdx.x;                              // b*NC + n
  int b = i >> 10, n = i & (NC - 1);
  int h = threadIdx.x;
  float m1 = divided[((size_t)(b * TT + t) * NC + n) * 3 + 0];
  if (m1 <= 0.f) {                                 // uniform per block
    if (h < HID) h_new[(size_t)i * HID + h] = 0.f;
    return;
  }
  __shared__ float co_s[GS];
  __shared__ float hp_s[HID];
  if (h < GS)  co_s[h] = co[(size_t)i * GS + h];
  if (h < HID) hp_s[h] = h_prev[(size_t)i * HID + h];
  __syncthreads();
  if (h >= HID) return;
  float ir = bi[h], iz = bi[HID + h], inn = bi[2 * HID + h];
  for (int c = 0; c < GS; ++c) {
    float x = co_s[c]; const float* w = Wi + c * 3 * HID;
    ir += x * w[h]; iz += x * w[HID + h]; inn += x * w[2 * HID + h];
  }
  float hr = bh[h], hz = bh[HID + h], hn = bh[2 * HID + h];
  for (int c = 0; c < HID; ++c) {
    float x = hp_s[c]; const float* w = Wh + c * 3 * HID;
    hr += x * w[h]; hz += x * w[HID + h]; hn += x * w[2 * HID + h];
  }
  float r  = 1.f / (1.f + __expf(-(ir + hr)));
  float z  = 1.f / (1.f + __expf(-(iz + hz)));
  float ng = tanhf(inn + r * hn);
  float hm1 = (1.f - z) * ng + z * hp_s[h];
  h_new[(size_t)i * HID + h] = hm1;
  atomicMax(&om1e[b * HP + h], encf(hm1));
}

// ---------------- flash attention over codes, bf16 WMMA, one wave per 16-row tile ----------------
__global__ void __launch_bounds__(32)
k_attn(const float* __restrict__ divided,
       const unsigned short* __restrict__ qb, const unsigned short* __restrict__ kb,
       const unsigned short* __restrict__ vtb,
       float* __restrict__ h_new, unsigned* __restrict__ om23e, int t) {
  int b = blockIdx.y;
  int i0 = blockIdx.x * 16;
  int lane = threadIdx.x;
  int half = lane >> 4, lm = lane & 15;
  const float* dvb = divided + (size_t)(b * TT + t) * NC * 3;

  float dmy = 0.f;
  if (lane < 16) dmy = dvb[(i0 + lane) * 3 + 1] + dvb[(i0 + lane) * 3 + 2];
  if (__ballot(dmy > 0.f) == 0) return;            // no active query row in this tile

  FragBF qa;                                        // A-frag: rows i0..i0+15, K=0..31
  {
    const unsigned short* qp = qb + ((size_t)b * NC + i0 + lm) * GS + half * 16;
    qa.q[0] = ((const uint4*)qp)[0];
    qa.q[1] = *(const uint4*)(qp + 8);
  }
  float mrow[8], lrow[8];
#pragma unroll
  for (int r = 0; r < 8; ++r) { mrow[r] = -__builtin_inff(); lrow[r] = 0.f; }
  v8f o[10] = {};
  __shared__ __align__(16) unsigned short Pst[16 * 32];
  const float scale = 0.17677669529663687f;         // 1/sqrt(32)

  for (int j0 = 0; j0 < NC; j0 += 32) {
    float s0[8], s1[8];
#pragma unroll
    for (int sub = 0; sub < 2; ++sub) {
      int j = j0 + sub * 16 + lm;
      FragBF kf;
      const unsigned short* kp = kb + ((size_t)b * NC + j) * GS + half * 16;
      kf.q[0] = ((const uint4*)kp)[0];
      kf.q[1] = ((const uint4*)kp)[1];
      v8f s = {};
      s = __builtin_amdgcn_wmma_f32_16x16x32_bf16(false, qa.v, false, kf.v,
                                                  (short)0, s, false, false);
      float m23j = dvb[j * 3 + 1] + dvb[j * 3 + 2];
      float* dst = sub ? s1 : s0;
#pragma unroll
      for (int r = 0; r < 8; ++r) dst[r] = (m23j > 0.f) ? s[r] * scale : NEGINF;
    }
    // online softmax update (rows live across 16 lanes of each half-wave)
#pragma unroll
    for (int r = 0; r < 8; ++r) {
      float rm = fmaxf(s0[r], s1[r]);
      rm = fmaxf(rm, __shfl_xor(rm, 1, 32));
      rm = fmaxf(rm, __shfl_xor(rm, 2, 32));
      rm = fmaxf(rm, __shfl_xor(rm, 4, 32));
      rm = fmaxf(rm, __shfl_xor(rm, 8, 32));
      float mnew = fmaxf(mrow[r], rm);
      float f  = __expf(mrow[r] - mnew);
      float p0 = __expf(s0[r] - mnew);
      float p1 = __expf(s1[r] - mnew);
      float rs = p0 + p1;
      rs += __shfl_xor(rs, 1, 32);
      rs += __shfl_xor(rs, 2, 32);
      rs += __shfl_xor(rs, 4, 32);
      rs += __shfl_xor(rs, 8, 32);
      lrow[r] = lrow[r] * f + rs;
      mrow[r] = mnew;
#pragma unroll
      for (int ht = 0; ht < 10; ++ht) o[ht][r] *= f;
      Pst[(r + half * 8) * 32 + lm]      = f2bf(p0);
      Pst[(r + half * 8) * 32 + 16 + lm] = f2bf(p1);
    }
    __syncthreads();
    FragBF pa;                                      // re-layout P as A-fragment
    {
      const unsigned short* pp = Pst + lm * 32 + half * 16;
      pa.q[0] = ((const uint4*)pp)[0];
      pa.q[1] = *(const uint4*)(pp + 8);
    }
#pragma unroll
    for (int ht = 0; ht < 10; ++ht) {
      FragBF vf;
      const unsigned short* vp = vtb + ((size_t)b * HP + ht * 16 + lm) * NC + j0 + half * 16;
      vf.q[0] = ((const uint4*)vp)[0];
      vf.q[1] = ((const uint4*)vp)[1];
      o[ht] = __builtin_amdgcn_wmma_f32_16x16x32_bf16(false, pa.v, false, vf.v,
                                                      (short)0, o[ht], false, false);
    }
    __syncthreads();
  }
#pragma unroll
  for (int r = 0; r < 8; ++r) {
    int row = i0 + r + half * 8;
    float dm = dvb[row * 3 + 1] + dvb[row * 3 + 2];
    if (dm <= 0.f) continue;
    float inv = 1.f / lrow[r];
#pragma unroll
    for (int ht = 0; ht < 10; ++ht) {
      int h = ht * 16 + lm;
      if (h >= HID) continue;
      float val = tanhf(o[ht][r] * inv);
      h_new[((size_t)b * NC + row) * HID + h] = val;
      atomicMax(&om23e[b * HP + h], encf(val));
    }
  }
}

// ---------------- per-step output: sum of subset max-pools ----------------
__global__ void k_stepout(const unsigned* __restrict__ om1e, const unsigned* __restrict__ om23e,
                          float* __restrict__ outs, int t) {
  int b = blockIdx.x, h = threadIdx.x;
  if (h >= HID) return;
  unsigned e1 = om1e[b * HP + h], e2 = om23e[b * HP + h];
  float v1 = e1 ? decf(e1) : 0.f;                  // 0 sentinel == empty subset
  float v2 = e2 ? decf(e2) : 0.f;
  outs[((size_t)b * TT + t) * HP + h] = v1 + v2;
}

// ---------------- visit attention pooling + classifier ----------------
__global__ void __launch_bounds__(256)
k_final(const float* __restrict__ outs, const int* __restrict__ lens,
        const float* __restrict__ Wd, const float* __restrict__ bd,
        const float* __restrict__ ctx, const float* __restrict__ Wc,
        const float* __restrict__ bc, float* __restrict__ out) {
  int b = blockIdx.x, tid = threadIdx.x;
  __shared__ float sc[TT];
  __shared__ float pooled[HID];
  if (tid < TT) {
    const float* orow = outs + ((size_t)b * TT + tid) * HP;
    float acc = 0.f;
    for (int a = 0; a < 32; ++a) {
      float da = bd[a];
      for (int h = 0; h < HID; ++h) da += orow[h] * Wd[h * 32 + a];
      acc += ctx[a] * da;
    }
    sc[tid] = acc;
  }
  __syncthreads();
  if (tid == 0) {
    int len = lens[b];
    float mx = -__builtin_inff();
    for (int u = 0; u < TT; ++u) if (u < len) mx = fmaxf(mx, sc[u]);
    float s = 0.f;
    for (int u = 0; u < TT; ++u) { float e = (u < len) ? __expf(sc[u] - mx) : 0.f; sc[u] = e; s += e; }
    float inv = 1.f / s;
    for (int u = 0; u < TT; ++u) sc[u] *= inv;
  }
  __syncthreads();
  if (tid < HID) {
    float p = 0.f;
    for (int u = 0; u < TT; ++u) p += outs[((size_t)b * TT + u) * HP + tid] * sc[u];
    pooled[tid] = p;
  }
  __syncthreads();
  for (int o = tid; o < OUTN; o += 256) {
    float acc = bc[o];
    for (int h = 0; h < HID; ++h) acc += pooled[h] * Wc[h * OUTN + o];
    out[(size_t)b * OUTN + o] = 1.f / (1.f + __expf(-acc));
  }
}

// ---------------- host orchestration ----------------
extern "C" void kernel_launch(void* const* d_in, const int* in_sizes, int n_in,
                              void* d_out, int out_size, void* d_ws, size_t ws_size,
                              hipStream_t stream) {
  const float* code_x   = (const float*)d_in[0];
  const float* divided  = (const float*)d_in[1];
  const float* neigh    = (const float*)d_in[2];
  const int*   lens     = (const int*)  d_in[3];
  const float* adj      = (const float*)d_in[4];
  const float* c_emb    = (const float*)d_in[5];
  const float* n_emb    = (const float*)d_in[6];
  const float* u_emb    = (const float*)d_in[7];
  const float* Wg = (const float*)d_in[8];  const float* bg = (const float*)d_in[9];
  const float* Wi = (const float*)d_in[10]; const float* bi = (const float*)d_in[11];
  const float* Wh = (const float*)d_in[12]; const float* bh = (const float*)d_in[13];
  const float* Wq = (const float*)d_in[14]; const float* bq = (const float*)d_in[15];
  const float* Wk = (const float*)d_in[16]; const float* bk = (const float*)d_in[17];
  const float* Wv = (const float*)d_in[18]; const float* bv = (const float*)d_in[19];
  const float* Wd = (const float*)d_in[20]; const float* bd = (const float*)d_in[21];
  const float* ctx= (const float*)d_in[22];
  const float* Wc = (const float*)d_in[23]; const float* bc = (const float*)d_in[24];
  float* out = (float*)d_out;
  (void)in_sizes; (void)n_in; (void)out_size; (void)ws_size;

  char* p = (char*)d_ws;
  auto carve = [&](size_t bytes) { char* r = p; p += (bytes + 255) & ~(size_t)255; return r; };
  unsigned short* adjb = (unsigned short*)carve((size_t)NC * NC * 2);      // bf16 adj
  unsigned short* cet  = (unsigned short*)carve((size_t)NCOL * NC * 2);    // [384,1024]
  unsigned short* net  = (unsigned short*)carve((size_t)NCOL * NC * 2);
  float* adj_ce = (float*)carve((size_t)NC * NCOL * 4);                    // [1024,384]
  float* adj_ne = (float*)carve((size_t)NC * NCOL * 4);
  float* co     = (float*)carve((size_t)BN * NC * GS * 4);
  float* nobuf0 = (float*)carve((size_t)BN * NC * GS * 4);
  float* nobuf1 = (float*)carve((size_t)BN * NC * GS * 4);
  float* hbuf0  = (float*)carve((size_t)BN * NC * HID * 4);
  float* hbuf1  = (float*)carve((size_t)BN * NC * HID * 4);
  unsigned short* qbuf = (unsigned short*)carve((size_t)BN * NC * GS * 2);
  unsigned short* kbuf = (unsigned short*)carve((size_t)BN * NC * GS * 2);
  unsigned short* vtb  = (unsigned short*)carve((size_t)BN * HP * NC * 2); // V transposed
  unsigned* om1e  = (unsigned*)carve((size_t)BN * HP * 4);
  unsigned* om23e = (unsigned*)carve((size_t)BN * HP * 4);
  float* outs = (float*)carve((size_t)BN * TT * HP * 4);

  k_zero<<<(BN * NC * HID + 255) / 256, 256, 0, stream>>>(hbuf0, BN * NC * HID);
  k_cvt_bf16<<<(NC * NC + 255) / 256, 256, 0, stream>>>(adj, adjb, NC * NC);

  for (int t = 0; t < TT; ++t) {
    float* h_prev  = (t & 1) ? hbuf1 : hbuf0;
    float* h_new   = (t & 1) ? hbuf0 : hbuf1;
    float* no_prev = (t & 1) ? nobuf1 : nobuf0;
    float* no_cur  = (t & 1) ? nobuf0 : nobuf1;

    k_clear_om<<<(BN * HP + 255) / 256, 256, 0, stream>>>(om1e, om23e);
    k_mask_embed<<<(NCOL * NC + 255) / 256, 256, 0, stream>>>(code_x, neigh, c_emb, n_emb,
                                                              cet, net, t);
    dim3 gg(NC / 16, NCOL / 64);
    k_gemm_adj<<<gg, 32, 0, stream>>>(adjb, cet, adj_ce);
    k_gemm_adj<<<gg, 32, 0, stream>>>(adjb, net, adj_ne);
    k_co_no<<<(BN * NC + 255) / 256, 256, 0, stream>>>(code_x, neigh, c_emb, n_emb,
                                                       adj_ce, adj_ne, Wg, bg, co, no_cur, t);
    if (t > 0)
      k_qkv<<<(BN * NC + 255) / 256, 256, 0, stream>>>(divided, no_prev, u_emb, co,
                                                       Wq, bq, Wk, bk, Wv, bv,
                                                       qbuf, kbuf, vtb, t);
    k_gru<<<BN * NC, HP, 0, stream>>>(divided, co, h_prev, Wi, bi, Wh, bh, h_new, om1e, t);
    if (t > 0)
      k_attn<<<dim3(NC / 16, BN), 32, 0, stream>>>(divided, qbuf, kbuf, vtb, h_new, om23e, t);
    k_stepout<<<BN, HP, 0, stream>>>(om1e, om23e, outs, t);
  }
  k_final<<<BN, 256, 0, stream>>>(outs, lens, Wd, bd, ctx, Wc, bc, out);
}
